// GatedDeltaNet_17343077941328
// MI455X (gfx1250) — matrix operand order repaired and checked by reading, not compiled, not run on hardware
//
#include <hip/hip_runtime.h>
#include <hip/hip_bf16.h>

// ---------------- problem constants (match reference) ----------------
#define BB 2
#define LL 4096
#define DD 2048
#define HH 4
#define DKk 256
#define DVv 512
#define CONVK 4

typedef __bf16    v16bf __attribute__((ext_vector_type(16)));
typedef __bf16    v8bf  __attribute__((ext_vector_type(8)));
typedef float     v8f   __attribute__((ext_vector_type(8)));
typedef float     f4    __attribute__((ext_vector_type(4)));
typedef unsigned  u4    __attribute__((ext_vector_type(4)));

// pack two f32 -> two bf16 in one dword:
// round-to-nearest bias then v_perm_b32 grabs the two high halves (3 VALU).
static __device__ inline unsigned pk2bf(float a, float b) {
    unsigned ra = __builtin_bit_cast(unsigned, a) + 0x8000u;
    unsigned rb = __builtin_bit_cast(unsigned, b) + 0x8000u;
    // dst = { rb[31:16], ra[31:16] } : bytes {7,6,3,2} of {rb,ra}
    return __builtin_amdgcn_perm(rb, ra, 0x07060302u);
}

// =====================================================================
// GEMM: C[M,N] = A[M,K] * W[K,N]   (fp32 memory, bf16 WMMA, f32 acc)
// Block: 256 threads = 8 waves. Tile: 256(M) x 64(N), K-step 32.
// Each wave computes two 16-row bands x 64 cols = 8 WMMAs per K-step
// on 2 A-fragments + 4 shared B-fragments (12 ds_load_b128).
//
// Double-buffered LDS, fragment-order layout:
//  - As[b][r][p] holds A[m0+r][kk + p]; lane-half hl reads the
//    contiguous 32 B at As[b][row][hl*16].
//  - Bst[b][c][k] is the transposed B tile; lane reads Bst[b][col][kb..kb+15].
// Row stride padded to 40 elems (80 B) -> conflict-free ds_load_b128.
// Pipeline: frag ds_loads -> next-tile global_load_b128 -> WMMA x8 ->
// v_perm pack + ds_store_b128 next tile -> barrier.
// =====================================================================
#define TM 256
#define TN 64
#define TK 32
#define TKP 40   // padded LDS row stride (elements)

template <int N>
__global__ __launch_bounds__(256)
void gemm_bf16_kernel(const float* __restrict__ A, const float* __restrict__ W,
                      float* __restrict__ C, int M, int K) {
    __shared__ __bf16 As[2][TM][TKP];    // 40 KB
    __shared__ __bf16 Bst[2][TN][TKP];   // 10 KB
    const int tid  = threadIdx.x;
    const int wave = tid >> 5;
    const int lane = tid & 31;
    const int m0   = blockIdx.y * TM;
    const int n0   = blockIdx.x * TN;

    // staging work assignment (per thread):
    //  A: 4 items of (row, 8-col group); B: 1 item of (col, 8-K group)
    const int bc  = tid & 63;
    const int br0 = (tid >> 6) * 8;

    f4 ga[4][2];
    float gb[8];

    auto gload = [&](int kk) {
        #pragma unroll
        for (int it = 0; it < 4; ++it) {
            int w = tid + it * 256;
            int row = w >> 2, c0 = (w & 3) * 8;
            const float* ap = &A[(size_t)(m0 + row) * K + kk + c0];
            ga[it][0] = *reinterpret_cast<const f4*>(ap);
            ga[it][1] = *reinterpret_cast<const f4*>(ap + 4);
            if (it == 0 && kk + TK < K)
                __builtin_prefetch(ap + TK, 0, 1);   // global_prefetch_b8
        }
        const float* wp = &W[(size_t)(kk + br0) * N + n0 + bc];
        #pragma unroll
        for (int i = 0; i < 8; ++i) gb[i] = wp[(size_t)i * N];
    };

    auto lstore = [&](int buf) {
        #pragma unroll
        for (int it = 0; it < 4; ++it) {
            int w = tid + it * 256;
            int row = w >> 2, c0 = (w & 3) * 8;
            int pbase = c0; // contiguous k order
            u4 pk;
            pk[0] = pk2bf(ga[it][0][0], ga[it][0][1]);
            pk[1] = pk2bf(ga[it][0][2], ga[it][0][3]);
            pk[2] = pk2bf(ga[it][1][0], ga[it][1][1]);
            pk[3] = pk2bf(ga[it][1][2], ga[it][1][3]);
            *reinterpret_cast<u4*>(&As[buf][row][pbase]) = pk;
        }
        u4 pb;
        #pragma unroll
        for (int i = 0; i < 4; ++i) pb[i] = pk2bf(gb[2 * i], gb[2 * i + 1]);
        *reinterpret_cast<u4*>(&Bst[buf][bc][br0]) = pb;
    };

    v8f acc[2][4] = {};
    const int arow0 = wave * 32 + (lane & 15);
    const int hl16  = (lane >> 4) * 16;
    const int bcol0 = lane & 15;
    const int kb    = (lane >> 4) * 16;

    gload(0);
    lstore(0);
    __syncthreads();

    int buf = 0;
    for (int kk = 0; kk < K; kk += TK, buf ^= 1) {
        // fragments from current buffer (ds_load_b128 x2 each)
        v16bf afrag[2];
        #pragma unroll
        for (int bd = 0; bd < 2; ++bd) {
            const __bf16* apf = &As[buf][arow0 + bd * 16][hl16];
            v8bf alo = *reinterpret_cast<const v8bf*>(apf);
            v8bf ahi = *reinterpret_cast<const v8bf*>(apf + 8);
            afrag[bd] = __builtin_shufflevector(alo, ahi,
                0,1,2,3,4,5,6,7,8,9,10,11,12,13,14,15);
        }
        v16bf bfr[4];
        #pragma unroll
        for (int nt = 0; nt < 4; ++nt) {
            const __bf16* bp = &Bst[buf][nt * 16 + bcol0][kb];
            v8bf blo = *reinterpret_cast<const v8bf*>(bp);
            v8bf bhi = *reinterpret_cast<const v8bf*>(bp + 8);
            bfr[nt] = __builtin_shufflevector(blo, bhi,
                0,1,2,3,4,5,6,7,8,9,10,11,12,13,14,15);
        }

        const bool more = (kk + TK) < K;
        if (more) gload(kk + TK);          // issue global loads for next tile

        #pragma unroll
        for (int nt = 0; nt < 4; ++nt) {
            acc[0][nt] = __builtin_amdgcn_wmma_f32_16x16x32_bf16(
                false, afrag[0], false, bfr[nt], (short)0, acc[0][nt], false, false);
            acc[1][nt] = __builtin_amdgcn_wmma_f32_16x16x32_bf16(
                false, afrag[1], false, bfr[nt], (short)0, acc[1][nt], false, false);
        }

        if (more) lstore(buf ^ 1);         // overlap pack/ds_store with WMMA pipe
        __syncthreads();
    }

    // C/D layout: VGPR r -> M = r (lanes 0-15) / M = 8+r (lanes 16-31); N = lane&15
    const int rbase = (lane >> 4) * 8;
    const int cl    = lane & 15;
    #pragma unroll
    for (int bd = 0; bd < 2; ++bd) {
        float* cp = C + (size_t)(m0 + wave * 32 + bd * 16 + rbase) * N + (n0 + cl);
        #pragma unroll
        for (int nt = 0; nt < 4; ++nt) {
            #pragma unroll
            for (int r = 0; r < 8; ++r)
                cp[(size_t)r * N + nt * 16] = acc[bd][nt][r];
        }
    }
}

// =====================================================================
// Causal depthwise conv (K=4) + SiLU over [B,L,C]
// =====================================================================
__global__ __launch_bounds__(256)
void conv_silu_kernel(const float* __restrict__ x, const float* __restrict__ w,
                      float* __restrict__ y, int Cc) {
    size_t idx = (size_t)blockIdx.x * 256 + threadIdx.x;
    size_t total = (size_t)BB * LL * Cc;
    if (idx >= total) return;
    int c = (int)(idx % Cc);
    size_t bl = idx / Cc;
    int l = (int)(bl % LL);
    int b = (int)(bl / LL);
    float s = 0.f;
    #pragma unroll
    for (int i = 0; i < CONVK; ++i) {
        int ll = l - (CONVK - 1) + i;
        if (ll >= 0)
            s += w[(size_t)c * CONVK + i] * x[((size_t)b * LL + ll) * Cc + c];
    }
    y[idx] = s / (1.f + __expf(-s));   // SiLU
}

// =====================================================================
// Per-token GEMVs: g = -exp(A_log) + softplus(h@Wa + dt_bias); beta = sigmoid(h@Wb)
// =====================================================================
__global__ __launch_bounds__(256)
void ab_kernel(const float* __restrict__ h, const float* __restrict__ Wa,
               const float* __restrict__ Wb, const float* __restrict__ A_log,
               const float* __restrict__ dt_bias,
               float* __restrict__ gvec, float* __restrict__ beta) {
    __shared__ float red[8][256];
    const size_t bl = blockIdx.x;
    const float* hp = h + bl * DD;
    const int tid = threadIdx.x;
    float pa[HH] = {}, pb[HH] = {};
    for (int d = tid; d < DD; d += 256) {
        float hv = hp[d];
        #pragma unroll
        for (int q = 0; q < HH; ++q) {
            pa[q] += hv * Wa[(size_t)d * HH + q];
            pb[q] += hv * Wb[(size_t)d * HH + q];
        }
    }
    #pragma unroll
    for (int q = 0; q < HH; ++q) { red[q][tid] = pa[q]; red[HH + q][tid] = pb[q]; }
    __syncthreads();
    for (int s = 128; s > 0; s >>= 1) {
        if (tid < s)
            #pragma unroll
            for (int r = 0; r < 8; ++r) red[r][tid] += red[r][tid + s];
        __syncthreads();
    }
    if (tid < HH) {
        float da = red[tid][0] + dt_bias[tid];
        float sp = (da > 20.f) ? da : log1pf(__expf(da));
        gvec[bl * HH + tid] = -__expf(A_log[tid]) + sp;
        float db = red[HH + tid][0];
        beta[bl * HH + tid] = 1.f / (1.f + __expf(-db));
    }
}

// =====================================================================
// L2-normalize q (scaled by DK^-0.5) and k per (b,l,h). One block per (b,l).
// =====================================================================
__global__ __launch_bounds__(256)
void qknorm_kernel(float* __restrict__ q, float* __restrict__ k) {
    __shared__ float red[256];
    __shared__ float scl[HH];
    const size_t bl = blockIdx.x;
    float* qp = q + bl * (size_t)(HH * DKk);
    float* kp = k + bl * (size_t)(HH * DKk);
    const int tid = threadIdx.x, hh = tid >> 6, t = tid & 63;
    const int base = hh * DKk + t * 4;
    float qv[4], kv[4], sq = 0.f, sk = 0.f;
    #pragma unroll
    for (int i = 0; i < 4; ++i) {
        qv[i] = qp[base + i]; sq += qv[i] * qv[i];
        kv[i] = kp[base + i]; sk += kv[i] * kv[i];
    }
    red[tid] = sq; __syncthreads();
    for (int s = 32; s > 0; s >>= 1) { if (t < s) red[tid] += red[tid + s]; __syncthreads(); }
    if (t == 0) scl[hh] = rsqrtf(red[tid] + 1e-6f) * 0.0625f;   // * DK^-0.5
    __syncthreads();
    const float qscale = scl[hh];
    red[tid] = sk; __syncthreads();
    for (int s = 32; s > 0; s >>= 1) { if (t < s) red[tid] += red[tid + s]; __syncthreads(); }
    if (t == 0) scl[hh] = rsqrtf(red[tid] + 1e-6f);
    __syncthreads();
    const float kscale = scl[hh];
    #pragma unroll
    for (int i = 0; i < 4; ++i) {
        qp[base + i] = qv[i] * qscale;
        kp[base + i] = kv[i] * kscale;
    }
}

// =====================================================================
// Gated delta-rule scan. One block per (b, h, DV-tile of 128 columns).
// State S[DK=256][128] fp32 lives in LDS (128 KB, dynamic).
// =====================================================================
__global__ __launch_bounds__(256)
void scan_kernel(const float* __restrict__ q, const float* __restrict__ k,
                 const float* __restrict__ v, const float* __restrict__ g,
                 const float* __restrict__ beta, float* __restrict__ o) {
    extern __shared__ float smem[];
    float* S   = smem;                 // 256*128
    float* qs  = S + DKk * 128;        // 256
    float* ks  = qs + DKk;             // 256
    float* par = ks + DKk;             // 256
    float* dl  = par + 256;            // 128

    const int blk = blockIdx.x;
    const int jt = blk & 3;            // DV tile
    const int hh = (blk >> 2) & 3;     // head
    const int b  = blk >> 4;           // batch
    const int tid = threadIdx.x;
    const int j  = tid & 127;
    const int dh = (tid >> 7) * 128;

    for (int i = tid; i < DKk * 128; i += 256) S[i] = 0.f;
    __syncthreads();

    for (int t = 0; t < LL; ++t) {
        const size_t qbase = (((size_t)b * LL + t) * HH + hh) * DKk;
        const size_t vbase = (((size_t)b * LL + t) * HH + hh) * DVv + (size_t)jt * 128;
        qs[tid] = q[qbase + tid];
        ks[tid] = k[qbase + tid];
        const float gt = g[((size_t)b * LL + t) * HH + hh];
        const float bt = beta[((size_t)b * LL + t) * HH + hh];
        const float decay = __expf(gt);
        const float vj = (tid < 128) ? v[vbase + tid] : 0.f;
        __syncthreads();

        float part = 0.f;
        #pragma unroll 4
        for (int d = 0; d < 128; ++d) part += ks[dh + d] * S[(dh + d) * 128 + j];
        par[tid] = part;
        __syncthreads();
        if (tid < 128) {
            float kS = decay * (par[tid] + par[tid + 128]);
            dl[tid] = bt * (vj - kS);
        }
        __syncthreads();

        const float del = dl[j];
        float opart = 0.f;
        #pragma unroll 4
        for (int d = 0; d < 128; ++d) {
            int dd = dh + d;
            float sv = S[dd * 128 + j] * decay + ks[dd] * del;
            S[dd * 128 + j] = sv;
            opart += qs[dd] * sv;
        }
        par[tid] = opart;
        __syncthreads();
        if (tid < 128) o[vbase + tid] = par[tid] + par[tid + 128];
        __syncthreads();
    }
}

// =====================================================================
// Gated RMSNorm: per (b,l,h): rms over DV, * norm_w, * silu(gate)
// =====================================================================
__global__ __launch_bounds__(256)
void rmsgate_kernel(const float* __restrict__ o, const float* __restrict__ gate,
                    const float* __restrict__ nw, float* __restrict__ out) {
    __shared__ float red[256];
    const size_t blh = blockIdx.x;
    const float* op = o + blh * DVv;
    const float* gp = gate + blh * DVv;
    float* yp = out + blh * DVv;
    const int tid = threadIdx.x;
    float x0 = op[tid], x1 = op[tid + 256];
    red[tid] = x0 * x0 + x1 * x1;
    __syncthreads();
    for (int s = 128; s > 0; s >>= 1) { if (tid < s) red[tid] += red[tid + s]; __syncthreads(); }
    const float r = rsqrtf(red[0] * (1.0f / DVv) + 1e-5f);
    float g0 = gp[tid], g1 = gp[tid + 256];
    yp[tid]       = x0 * r * nw[tid]       * (g0 / (1.f + __expf(-g0)));
    yp[tid + 256] = x1 * r * nw[tid + 256] * (g1 / (1.f + __expf(-g1)));
}

// =====================================================================
extern "C" void kernel_launch(void* const* d_in, const int* in_sizes, int n_in,
                              void* d_out, int out_size, void* d_ws, size_t ws_size,
                              hipStream_t stream) {
    (void)in_sizes; (void)n_in; (void)out_size; (void)ws_size;
    const float* h       = (const float*)d_in[0];
    const float* Wq      = (const float*)d_in[1];
    const float* Wk      = (const float*)d_in[2];
    const float* Wv      = (const float*)d_in[3];
    const float* Wa      = (const float*)d_in[4];
    const float* Wb      = (const float*)d_in[5];
    const float* Wg      = (const float*)d_in[6];
    const float* Wo      = (const float*)d_in[7];
    const float* A_log   = (const float*)d_in[8];
    const float* dt_bias = (const float*)d_in[9];
    const float* conv_q  = (const float*)d_in[10];
    const float* conv_k  = (const float*)d_in[11];
    const float* conv_v  = (const float*)d_in[12];
    const float* norm_w  = (const float*)d_in[13];
    float* out = (float*)d_out;

    const size_t M = (size_t)BB * LL;            // 8192 tokens
    float* ws = (float*)d_ws;
    size_t off = 0;
    float* tmp   = ws + off; off += M * 2048;    // widest intermediate, reused
    float* qb    = ws + off; off += M * 1024;
    float* kb    = ws + off; off += M * 1024;
    float* vb    = ws + off; off += M * 2048;
    float* gateb = ws + off; off += M * 2048;
    float* gvec  = ws + off; off += M * HH;
    float* betab = ws + off; off += M * HH;
    float* ob    = ws + off;

    const dim3 blk(256);
    const dim3 g1024(1024 / TN, (unsigned)(M / TM));
    const dim3 g2048(2048 / TN, (unsigned)(M / TM));

    // q = silu(conv(h@Wq))
    gemm_bf16_kernel<1024><<<g1024, blk, 0, stream>>>(h, Wq, tmp, (int)M, DD);
    conv_silu_kernel<<<(unsigned)((M * 1024 + 255) / 256), blk, 0, stream>>>(tmp, conv_q, qb, 1024);
    // k
    gemm_bf16_kernel<1024><<<g1024, blk, 0, stream>>>(h, Wk, tmp, (int)M, DD);
    conv_silu_kernel<<<(unsigned)((M * 1024 + 255) / 256), blk, 0, stream>>>(tmp, conv_k, kb, 1024);
    // v
    gemm_bf16_kernel<2048><<<g2048, blk, 0, stream>>>(h, Wv, tmp, (int)M, DD);
    conv_silu_kernel<<<(unsigned)((M * 2048 + 255) / 256), blk, 0, stream>>>(tmp, conv_v, vb, 2048);
    // gate = h@Wg
    gemm_bf16_kernel<2048><<<g2048, blk, 0, stream>>>(h, Wg, gateb, (int)M, DD);
    // g, beta
    ab_kernel<<<(unsigned)M, blk, 0, stream>>>(h, Wa, Wb, A_log, dt_bias, gvec, betab);
    // q/k L2 norm (+ q * DK^-0.5)
    qknorm_kernel<<<(unsigned)M, blk, 0, stream>>>(qb, kb);
    // gated delta-rule scan: 2*4*4 = 32 blocks, ~131.5 KB dynamic LDS each
    const size_t scan_lds = (size_t)(DKk * 128 + 3 * 256 + 128) * sizeof(float);
    scan_kernel<<<BB * HH * (DVv / 128), blk, scan_lds, stream>>>(qb, kb, vb, gvec, betab, ob);
    // gated RMSNorm -> tmp (reused as "normed")
    rmsgate_kernel<<<(unsigned)(M * HH), blk, 0, stream>>>(ob, gateb, norm_w, tmp);
    // out = normed @ Wo
    gemm_bf16_kernel<2048><<<g2048, blk, 0, stream>>>(tmp, Wo, out, (int)M, DD);
}
